// MultiHeadSelfAttention_32667521254190
// MI455X (gfx1250) — matrix operand order repaired and checked by reading, not compiled, not run on hardware
//
#include <hip/hip_runtime.h>
#include <hip/hip_bf16.h>
#include <math.h>
#include <stdint.h>

// ---------------------------------------------------------------------------
// MHA forward for MI455X (gfx1250, wave32, WMMA + async-to-LDS staging).
// B=4, S=2048, D=1024, H=16, DK=64. bf16 WMMA with f32 accumulation.
// ---------------------------------------------------------------------------

#define BB   4
#define SS   2048
#define DD   1024
#define HH   16
#define DK   64

typedef __attribute__((ext_vector_type(16))) __bf16 bf16x16;
typedef __attribute__((ext_vector_type(8)))  __bf16 bf16x8;
typedef __attribute__((ext_vector_type(8)))  float  f32x8;

union BF16x16U { bf16x16 v; bf16x8 h[2]; };

static __device__ inline bf16x16 ld2(const __bf16* p0, const __bf16* p1) {
    BF16x16U u;
    u.h[0] = *(const bf16x8*)p0;
    u.h[1] = *(const bf16x8*)p1;
    return u.v;
}

static __device__ inline f32x8 wmma_bf16(bf16x16 a, bf16x16 b, f32x8 c) {
    return __builtin_amdgcn_wmma_f32_16x16x32_bf16(
        /*neg_a=*/false, a, /*neg_b=*/false, b,
        /*c_mod=*/(short)0, c, /*reuse_a=*/false, /*reuse_b=*/false);
}

// CDNA5 async global->LDS copy (ASYNCcnt-tracked, no VGPR staging).
static __device__ inline void async_copy_b128(uint32_t lds_off, const void* gsrc) {
    asm volatile("global_load_async_to_lds_b128 %0, %1, off"
                 :: "v"(lds_off), "v"(gsrc) : "memory");
}
static __device__ inline void wait_async0() {
#if __has_builtin(__builtin_amdgcn_s_wait_asynccnt)
    __builtin_amdgcn_s_wait_asynccnt(0);
#else
    asm volatile("s_wait_asynccnt 0" ::: "memory");
#endif
}

// ---------------------------------------------------------------------------
// Elementwise casts
// ---------------------------------------------------------------------------
__global__ void cast_f32_bf16(const float* __restrict__ in, __bf16* __restrict__ out, int n) {
    int i = blockIdx.x * blockDim.x + threadIdx.x;
    if (i < n) out[i] = (__bf16)in[i];
}

// W[k][n] f32 -> Wt[n][k] bf16  (so GEMM B tiles are contiguous along K)
__global__ void transpose_cast_w(const float* __restrict__ W, __bf16* __restrict__ Wt) {
    int t = blockIdx.x * blockDim.x + threadIdx.x;
    if (t >= DD * DD) return;
    int n = t >> 10;
    int k = t & (DD - 1);
    Wt[t] = (__bf16)W[(size_t)k * DD + n];
}

// ---------------------------------------------------------------------------
// RoPE in-place on [b,h,s,d] bf16 buffer (per even/odd pair)
// ---------------------------------------------------------------------------
__global__ void rope_inplace(__bf16* __restrict__ T, int npairs) {
    int t = blockIdx.x * blockDim.x + threadIdx.x;
    if (t >= npairs) return;
    int d2  = t & 31;
    int row = t >> 5;            // (b*H + h)*S + s
    int s   = row & (SS - 1);
    __bf16* p = T + (size_t)row * DK + d2 * 2;
    float e = (float)p[0], o = (float)p[1];
    float ang = (float)s * powf(10000.0f, -(float)d2 * (1.0f / 32.0f));
    float sn, cs;
    sincosf(ang, &sn, &cs);
    p[0] = (__bf16)(e * cs - o * sn);
    p[1] = (__bf16)(o * cs + e * sn);
}

// ---------------------------------------------------------------------------
// GEMM: C[M,1024] = A[M,1024](bf16) x Bt[1024,1024](bf16, B^T row-major)
// One wave computes a 32x64 tile (2 M-subtiles x 4 N-accumulators): the 4
// B-tiles per K-chunk are reused by both M-subtiles -> 8 wmma / 12 b128.
// mode 0: bf16 out, layout [b,h,s,d], scaled          (Q, K projections)
// mode 1: bf16 out, layout [b,h,d,s], scaled          (V projection, transposed)
// mode 2: f32  out, row-major [M,1024]                (final O*Wo)
// ---------------------------------------------------------------------------
__global__ __launch_bounds__(256) void gemm_bf16_32x64(
    const __bf16* __restrict__ A, const __bf16* __restrict__ Bt,
    void* __restrict__ Cout, float scale, int mode)
{
    const int K = DD;
    int lane = threadIdx.x & 31, wave = threadIdx.x >> 5;
    int tile = blockIdx.x * 8 + wave;
    int m0 = (tile >> 4) * 32;          // 16 N-tiles of 64
    int n0 = (tile & 15) * 64;
    int half = lane >> 4, l16 = lane & 15;

    const __bf16* arow0 = A + (size_t)(m0 + l16) * K;
    const __bf16* arow1 = A + (size_t)(m0 + 16 + l16) * K;
    const __bf16* brow[4];
    #pragma unroll
    for (int g = 0; g < 4; ++g)
        brow[g] = Bt + (size_t)(n0 + g * 16 + l16) * K + half * 8;

    f32x8 z = {0.f,0.f,0.f,0.f,0.f,0.f,0.f,0.f};
    f32x8 acc[2][4] = {{z,z,z,z},{z,z,z,z}};

    for (int kk = 0; kk < K; kk += 32) {
        bf16x16 va0 = ld2(arow0 + kk + half * 8, arow0 + kk + 16 + half * 8);
        bf16x16 va1 = ld2(arow1 + kk + half * 8, arow1 + kk + 16 + half * 8);
        #pragma unroll
        for (int g = 0; g < 4; ++g) {
            bf16x16 vb = ld2(brow[g] + kk, brow[g] + kk + 16);
            acc[0][g] = wmma_bf16(va0, vb, acc[0][g]);
            acc[1][g] = wmma_bf16(va1, vb, acc[1][g]);
        }
    }

    #pragma unroll
    for (int mi = 0; mi < 2; ++mi) {
        int m_r0 = m0 + mi * 16 + half * 8;   // C row = r + 8*half
        if (mode == 2) {
            float* out = (float*)Cout;
            #pragma unroll
            for (int g = 0; g < 4; ++g) {
                int n = n0 + g * 16 + l16;
                #pragma unroll
                for (int r = 0; r < 8; ++r)
                    out[(size_t)(m_r0 + r) * DD + n] = acc[mi][g][r];
            }
        } else {
            __bf16* out = (__bf16*)Cout;
            #pragma unroll
            for (int g = 0; g < 4; ++g) {
                int n = n0 + g * 16 + l16;
                int h = n >> 6, d = n & (DK - 1);
                #pragma unroll
                for (int r = 0; r < 8; ++r) {
                    int m = m_r0 + r;
                    int b = m >> 11, s = m & (SS - 1);
                    size_t idx;
                    if (mode == 0)
                        idx = ((size_t)((b * HH + h) * SS + s)) * DK + d;     // [b,h,s,d]
                    else
                        idx = ((size_t)((b * HH + h) * DK + d)) * SS + s;     // [b,h,d,s]
                    out[idx] = (__bf16)(acc[mi][g][r] * scale);
                }
            }
        }
    }
}

// ---------------------------------------------------------------------------
// Flash attention: 8 waves per block handle 8 consecutive 16-query tiles of
// the SAME (b,h). Each 32-key K/V block is staged ONCE per block into LDS via
// global_load_async_to_lds_b128 (ASYNCcnt), then all waves feed WMMAs from LDS.
//
// Scores computed transposed: S^T = K_tile(A: 16 keys x 32dk) * Q^T(B)
//   -> C layout has q per-lane (N), keys across regs/halves (M).
// Softmax stats per-lane (xor-16 combine). P lands directly in A layout
// for O = P(A: 16q x 32keys) * V(B: 32keys x 16d), V staged from Vt[b,h,d,s].
// ---------------------------------------------------------------------------
__global__ __launch_bounds__(256) void attn_fwd(
    const __bf16* __restrict__ Q, const __bf16* __restrict__ Kc,
    const __bf16* __restrict__ Vt, __bf16* __restrict__ O)
{
    __shared__ __bf16 Ksm[32 * 64];   // [key 0..31][d 0..63]   4 KB
    __shared__ __bf16 Vsm[64 * 32];   // [d 0..63][key 0..31]   4 KB

    int lane = threadIdx.x & 31, wave = threadIdx.x >> 5;
    int tid  = threadIdx.x;
    int tile = blockIdx.x * 8 + wave;
    int bh = tile >> 7;                 // 128 q-tiles per (b,h); blocks never straddle bh
    int qt = tile & 127;
    int q_base = qt * 16;
    int half = lane >> 4, l16 = lane & 15;

    const __bf16* Qb = Q  + (size_t)bh * SS * DK;
    const __bf16* Kb = Kc + (size_t)bh * SS * DK;
    const __bf16* Vb = Vt + (size_t)bh * DK * SS;

    // Q tile as two B-matrices (dk chunks 0..31, 32..63): lane = q column
    bf16x16 qB[2];
    #pragma unroll
    for (int kg = 0; kg < 2; ++kg) {
        const __bf16* p = Qb + (size_t)(q_base + l16) * DK + kg * 32 + half * 8;
        qB[kg] = ld2(p, p + 16);
    }

    f32x8 z = {0.f,0.f,0.f,0.f,0.f,0.f,0.f,0.f};
    f32x8 oacc[4] = {z, z, z, z};
    float m_run = -INFINITY, l_run = 0.f;

    uint32_t ksm_base = (uint32_t)(uintptr_t)&Ksm[0];
    uint32_t vsm_base = (uint32_t)(uintptr_t)&Vsm[0];

    // Block-uniform loop bound: last wave's causal range
    int jmax = ((blockIdx.x * 8 + 7) & 127) * 16 + 16;

    for (int j = 0; j < jmax; j += 32) {
        // ---- cooperative async stage of K (32x64) and V^T (64x32) into LDS
        {
            // K rows j..j+31 are contiguous: 4 KB -> 256 x 16B
            const __bf16* ksrc = Kb + (size_t)j * DK + tid * 8;
            async_copy_b128(ksm_base + (uint32_t)tid * 16u, (const void*)ksrc);
            // V: 64 rows of Vt, 32 keys each: thread t -> row t/4, chunk t%4
            int d = tid >> 2, c = tid & 3;
            const __bf16* vsrc = Vb + (size_t)d * SS + j + c * 8;
            async_copy_b128(vsm_base + (uint32_t)(d * 32 + c * 8) * 2u, (const void*)vsrc);
        }
        wait_async0();
        __syncthreads();

        if (j < q_base + 16) {   // wave-uniform: EXEC stays all-ones for WMMA
            // ---- scores S^T for keys j..j+31 (two 16-key tiles) from LDS
            f32x8 st[2];
            #pragma unroll
            for (int kt = 0; kt < 2; ++kt) {
                f32x8 c = z;
                #pragma unroll
                for (int kg = 0; kg < 2; ++kg) {
                    const __bf16* p = &Ksm[(kt * 16 + l16) * 64 + kg * 32 + half * 8];
                    bf16x16 ka = ld2(p, p + 16);
                    c = wmma_bf16(ka, qB[kg], c);
                }
                st[kt] = c;
            }

            // ---- causal mask (only on diagonal-crossing blocks)
            if (j + 31 > q_base) {
                int q = q_base + l16;
                #pragma unroll
                for (int kt = 0; kt < 2; ++kt)
                    #pragma unroll
                    for (int r = 0; r < 8; ++r) {
                        int key = j + kt * 16 + half * 8 + r;
                        if (key > q) st[kt][r] = -INFINITY;
                    }
            }

            // ---- online softmax stats (per-lane q, combine halves via xor-16)
            float mb = -INFINITY;
            #pragma unroll
            for (int kt = 0; kt < 2; ++kt)
                #pragma unroll
                for (int r = 0; r < 8; ++r) mb = fmaxf(mb, st[kt][r]);
            mb = fmaxf(mb, __shfl_xor(mb, 16, 32));
            float m_new = fmaxf(m_run, mb);
            float alpha = __expf(m_run - m_new);

            float pf[16];
            float lb = 0.f;
            #pragma unroll
            for (int kt = 0; kt < 2; ++kt)
                #pragma unroll
                for (int r = 0; r < 8; ++r) {
                    float pe = __expf(st[kt][r] - m_new);
                    pf[kt * 8 + r] = pe;
                    lb += pe;
                }
            lb += __shfl_xor(lb, 16, 32);
            l_run = l_run * alpha + lb;
            m_run = m_new;

            // ---- pack P into A-matrix layout (matches per-lane key sets)
            BF16x16U pu;
            #pragma unroll
            for (int i = 0; i < 16; ++i) pu.v[i] = (__bf16)pf[i];

            // ---- rescale O accumulators: row q = q_base + r + 8*half
            #pragma unroll
            for (int r = 0; r < 8; ++r) {
                float ar = __shfl(alpha, half * 8 + r, 32);
                #pragma unroll
                for (int g = 0; g < 4; ++g) oacc[g][r] *= ar;
            }

            // ---- O += P * V  (V B-tiles from LDS: lane = d column, keys contiguous)
            #pragma unroll
            for (int g = 0; g < 4; ++g) {
                const __bf16* p = &Vsm[(g * 16 + l16) * 32 + half * 8];
                bf16x16 vb = ld2(p, p + 16);
                oacc[g] = wmma_bf16(pu.v, vb, oacc[g]);
            }
        }
        __syncthreads();
    }

    // ---- normalize and store to Ob [b, s, h*64+d] (bf16, row-major [B*S, D])
    float inv_l = 1.0f / l_run;
    int b = bh >> 4, h = bh & 15;
    #pragma unroll
    for (int r = 0; r < 8; ++r) {
        float il = __shfl(inv_l, half * 8 + r, 32);
        int s = q_base + half * 8 + r;
        size_t rowoff = ((size_t)(b * SS + s)) * DD + h * DK;
        #pragma unroll
        for (int g = 0; g < 4; ++g)
            O[rowoff + g * 16 + l16] = (__bf16)(oacc[g][r] * il);
    }
}

// ---------------------------------------------------------------------------
// Host-side launcher
// ---------------------------------------------------------------------------
extern "C" void kernel_launch(void* const* d_in, const int* in_sizes, int n_in,
                              void* d_out, int out_size, void* d_ws, size_t ws_size,
                              hipStream_t stream) {
    const float* x  = (const float*)d_in[0];
    const float* Wq = (const float*)d_in[1];
    const float* Wk = (const float*)d_in[2];
    const float* Wv = (const float*)d_in[3];
    const float* Wo = (const float*)d_in[4];

    char* w = (char*)d_ws;
    const size_t SZ_XB = (size_t)BB * SS * DD * 2;      // 16 MB
    const size_t SZ_W  = (size_t)DD * DD * 2;           //  2 MB
    const size_t SZ_H  = (size_t)BB * HH * SS * DK * 2; // 16 MB

    __bf16* xb  = (__bf16*)(w);
    __bf16* Wqt = (__bf16*)(w + SZ_XB);
    __bf16* Wkt = (__bf16*)(w + SZ_XB + SZ_W);
    __bf16* Wvt = (__bf16*)(w + SZ_XB + 2 * SZ_W);
    __bf16* Wot = (__bf16*)(w + SZ_XB + 3 * SZ_W);
    __bf16* Qb  = (__bf16*)(w + SZ_XB + 4 * SZ_W);
    __bf16* Kb  = (__bf16*)(w + SZ_XB + 4 * SZ_W + SZ_H);
    __bf16* Vt  = (__bf16*)(w + SZ_XB + 4 * SZ_W + 2 * SZ_H);
    __bf16* Ob  = (__bf16*)(w + SZ_XB + 4 * SZ_W + 3 * SZ_H);

    // 1) casts
    {
        int n = BB * SS * DD;
        cast_f32_bf16<<<(n + 255) / 256, 256, 0, stream>>>(x, xb, n);
        int nw = DD * DD;
        transpose_cast_w<<<(nw + 255) / 256, 256, 0, stream>>>(Wq, Wqt);
        transpose_cast_w<<<(nw + 255) / 256, 256, 0, stream>>>(Wk, Wkt);
        transpose_cast_w<<<(nw + 255) / 256, 256, 0, stream>>>(Wv, Wvt);
        transpose_cast_w<<<(nw + 255) / 256, 256, 0, stream>>>(Wo, Wot);
    }

    // 2) QKV projections: tiles = (8192/32)*(1024/64) = 4096 -> 512 blocks of 8 waves
    const int GEMM_BLOCKS = (BB * SS / 32) * (DD / 64) / 8;
    const float qscale = 0.125f;  // 1/sqrt(64); commutes with RoPE rotation
    gemm_bf16_32x64<<<GEMM_BLOCKS, 256, 0, stream>>>(xb, Wqt, Qb, qscale, 0);
    gemm_bf16_32x64<<<GEMM_BLOCKS, 256, 0, stream>>>(xb, Wkt, Kb, 1.0f,   0);
    gemm_bf16_32x64<<<GEMM_BLOCKS, 256, 0, stream>>>(xb, Wvt, Vt, 1.0f,   1);

    // 3) RoPE on Q, K
    {
        int npairs = BB * HH * SS * (DK / 2);
        rope_inplace<<<(npairs + 255) / 256, 256, 0, stream>>>(Qb, npairs);
        rope_inplace<<<(npairs + 255) / 256, 256, 0, stream>>>(Kb, npairs);
    }

    // 4) attention: B*H*(S/16) = 8192 wave-tiles -> 1024 blocks of 8 waves
    attn_fwd<<<(BB * HH * (SS / 16)) / 8, 256, 0, stream>>>(Qb, Kb, Vt, Ob);

    // 5) output projection -> f32 d_out
    gemm_bf16_32x64<<<GEMM_BLOCKS, 256, 0, stream>>>(Ob, Wot, d_out, 1.0f, 2);
}
